// RowAttentionBlock_44727789420924
// MI455X (gfx1250) — matrix operand order repaired and checked by reading, not compiled, not run on hardware
//
#include <hip/hip_runtime.h>
#include <hip/hip_bf16.h>

#define B_   2
#define I_   1024
#define J_   1024
#define D_   512
#define P_   64
#define H_   8
#define C_   32
#define HC_  256
#define S_   4              // J splits for the fused attention kernel
#define JSEG (J_ / S_)      // 256
#define INF_ 1.0e9f
#define EPS_ 1e-5f

typedef __attribute__((ext_vector_type(16))) _Float16 v16h;
typedef __attribute__((ext_vector_type(8)))  _Float16 v8h;
typedef __attribute__((ext_vector_type(8)))  float    v8f;

// ---------------------------------------------------------------------------
// WMMA fragment loaders per CDNA5 ISA 7.12.2 (wave32); every lane reads
// CONTIGUOUS memory (b128-mergeable).
// A [16(M) x 32(K)] f16 row-major: lane (m=lane&15, hf=lane>>4) holds
//   K = [8*hf .. 8*hf+7] and [16+8*hf .. 16+8*hf+7].
// B [32(K) x 16(N)] f16 from an N-MAJOR source (src[n*ld + k]): lane n holds
//   K = [8*hf .. 8*hf+7] and [16+8*hf .. 16+8*hf+7].
// C/D [16x16] f32: VGPR r -> row r + 8*hf, col = lane&15.
// ---------------------------------------------------------------------------
static __device__ __forceinline__ v16h load_a16(const _Float16* __restrict__ src, int ld) {
  const int lane = threadIdx.x & 31;
  const int m = lane & 15, hf = lane >> 4;
  const v8h lo = *(const v8h*)(src + (size_t)m * ld + 8 * hf);
  const v8h hi = *(const v8h*)(src + (size_t)m * ld + 16 + 8 * hf);
  v16h a;
#pragma unroll
  for (int e = 0; e < 8; ++e) { a[e] = lo[e]; a[8 + e] = hi[e]; }
  return a;
}

static __device__ __forceinline__ v16h load_bn16(const _Float16* __restrict__ src, int ld) {
  const int lane = threadIdx.x & 31;
  const int n = lane & 15, hf = lane >> 4;
  const v8h lo = *(const v8h*)(src + (size_t)n * ld + 8 * hf);
  const v8h hi = *(const v8h*)(src + (size_t)n * ld + 16 + 8 * hf);
  v16h b;
#pragma unroll
  for (int e = 0; e < 8; ++e) { b[e] = lo[e]; b[8 + e] = hi[e]; }
  return b;
}

static __device__ __forceinline__ v8f wmma_f16(v16h a, v16h b, v8f c) {
  return __builtin_amdgcn_wmma_f32_16x16x32_f16(false, a, false, b, (short)0, c,
                                                false, false);
}

static __device__ __forceinline__ float fast_sigmoid(float x) {
  return __builtin_amdgcn_rcpf(1.f + __expf(-x));   // v_rcp_f32, no Newton div
}

// ---------------------------------------------------------------------------
// Kernel 0a: fp32 [K][N] -> f16 TRANSPOSED [N][K] weight conversion (one-shot)
// ---------------------------------------------------------------------------
__global__ void cvtT_f16_kernel(const float* __restrict__ src, _Float16* __restrict__ dst,
                                int K, int N) {
  const int t = blockIdx.x * blockDim.x + threadIdx.x;
  if (t >= K * N) return;
  const int n = t % N, k = t / N;
  dst[(size_t)n * K + k] = (_Float16)src[t];
}

// ---------------------------------------------------------------------------
// Kernel 0b: pair-projection prep (gamma-folded stacked [Wp|Wpg], LN hoist)
// ---------------------------------------------------------------------------
__global__ void prep_pair_kernel(const float* __restrict__ Wp, const float* __restrict__ bp,
                                 const float* __restrict__ Wpg, const float* __restrict__ bpg,
                                 const float* __restrict__ lng, const float* __restrict__ lnb,
                                 _Float16* __restrict__ wpairT, float* __restrict__ csums) {
  for (int t = threadIdx.x; t < 16 * P_; t += blockDim.x) {
    const int n = t / P_, p = t % P_;
    const float wv = (n < 8) ? Wp[p * H_ + n] : Wpg[p * H_ + n - 8];
    wpairT[n * P_ + p] = (_Float16)(lng[p] * wv);
  }
  if (threadIdx.x < 32) {
    const int n = threadIdx.x & 15;
    const bool isb = threadIdx.x >= 16;
    float acc = 0.f;
    for (int p = 0; p < P_; ++p) {
      const float wv = (n < 8) ? Wp[p * H_ + n] : Wpg[p * H_ + n - 8];
      acc += (isb ? lnb[p] : lng[p]) * wv;
    }
    if (isb) acc += (n < 8) ? bp[n] : bpg[n - 8];
    csums[(isb ? 16 : 0) + n] = acc;
  }
}

// ---------------------------------------------------------------------------
// Kernel 1: LayerNorm of node_i / node_j; one wave per row of 512.
// ---------------------------------------------------------------------------
__global__ __launch_bounds__(256) void ln_nodes_kernel(
    const float* __restrict__ xi, const float* __restrict__ xj,
    const float* __restrict__ gi, const float* __restrict__ bi,
    const float* __restrict__ gj, const float* __restrict__ bj,
    float* __restrict__ ni32, _Float16* __restrict__ ni16,
    _Float16* __restrict__ nj16) {
  const int wave = (blockIdx.x * blockDim.x + threadIdx.x) >> 5;
  const int lane = threadIdx.x & 31;
  const int NI = B_ * I_;
  const float *x, *g, *bt;
  int row;
  bool is_i;
  if (wave < NI) { row = wave;      x = xi; g = gi; bt = bi; is_i = true;  }
  else           { row = wave - NI; x = xj; g = gj; bt = bj; is_i = false; }
  const float* xr = x + (size_t)row * D_;
  float vals[D_ / 32];
  float s = 0.f, s2 = 0.f;
#pragma unroll
  for (int e = 0; e < D_ / 32; ++e) {
    float v = xr[lane + 32 * e];
    vals[e] = v; s += v; s2 += v * v;
  }
#pragma unroll
  for (int m = 16; m >= 1; m >>= 1) {
    s  += __shfl_xor(s,  m, 32);
    s2 += __shfl_xor(s2, m, 32);
  }
  const float mean = s * (1.0f / D_);
  const float var  = s2 * (1.0f / D_) - mean * mean;
  const float rstd = rsqrtf(var + EPS_);
#pragma unroll
  for (int e = 0; e < D_ / 32; ++e) {
    const int col = lane + 32 * e;
    const float y = (vals[e] - mean) * rstd * g[col] + bt[col];
    if (is_i) {
      ni32[(size_t)row * D_ + col] = y;
      ni16[(size_t)row * D_ + col] = (_Float16)y;
    } else {
      nj16[(size_t)row * D_ + col] = (_Float16)y;
    }
  }
}

// ---------------------------------------------------------------------------
// Kernel 2: q/k/v/g projections, f16 WMMA GEMM with n-major W^T.
// V stored transposed [B][H][C][J] so the attention P·V operand is n-major.
// ---------------------------------------------------------------------------
__global__ __launch_bounds__(256) void qkvg_gemm_kernel(
    const _Float16* __restrict__ ni16, const _Float16* __restrict__ nj16,
    const _Float16* __restrict__ wT16, const float* __restrict__ bg,
    _Float16* __restrict__ out, _Float16* __restrict__ vT) {
  const int mat  = blockIdx.y;               // 0=q 1=k 2=v 3=g
  const int wave = (blockIdx.x * blockDim.x + threadIdx.x) >> 5;
  const int ntl  = HC_ / 16;
  const int m0   = (wave / ntl) * 16;
  const int n0   = (wave % ntl) * 16;
  const _Float16* in16 = (mat == 1 || mat == 2) ? nj16 : ni16;
  const _Float16* wT   = wT16 + (size_t)mat * D_ * HC_;
  v8f acc = {};
#pragma unroll 4
  for (int k0 = 0; k0 < D_; k0 += 32) {
    v16h a = load_a16(in16 + (size_t)m0 * D_ + k0, D_);
    v16h b = load_bn16(wT + (size_t)n0 * D_ + k0, D_);
    acc = wmma_f16(a, b, acc);
  }
  const int lane = threadIdx.x & 31;
  const int col = lane & 15, hf = lane >> 4;
  const float qscale = 0.1767766952966369f;  // 1/sqrt(32)
  if (mat == 2) {
    const int hc = n0 + col;
    const int h = hc >> 5, c = hc & 31;
#pragma unroll
    for (int r = 0; r < 8; ++r) {
      const int row = m0 + r + 8 * hf;
      const int b = row >> 10, j = row & (J_ - 1);
      vT[(((size_t)(b * H_ + h)) * C_ + c) * J_ + j] = (_Float16)acc[r];
    }
  } else {
    _Float16* o = out + (size_t)mat * (size_t)(B_ * I_) * HC_;
#pragma unroll
    for (int r = 0; r < 8; ++r) {
      const int row = m0 + r + 8 * hf;
      float v = acc[r];
      if (mat == 0)      v *= qscale;
      else if (mat == 3) v = fast_sigmoid(v + bg[n0 + col]);
      o[(size_t)row * HC_ + n0 + col] = (_Float16)v;
    }
  }
}

// ---------------------------------------------------------------------------
// Kernel 3: FUSED pair-bias + flash attention, SPLIT-J for occupancy.
// Grid = (I/16, B, S): block = 8 waves = 8 heads over one 16-row i-tile and a
// J/S segment. Per 32-j step:
//   stage 1: block computes bias tile [8h][16i][32j] in LDS (double-buffered)
//            via WMMA on raw-pe f16 with gamma-folded weights + LN hoist.
//   stage 2: each wave: score WMMA + online softmax + P·V WMMA.
// Emits partial (O, m, l) per split; merged by attn_merge_kernel.
// pe is still read exactly once across all splits.
// ---------------------------------------------------------------------------
__global__ __launch_bounds__(256) void attn_fused_kernel(
    const _Float16* __restrict__ q16, const _Float16* __restrict__ k16,
    const _Float16* __restrict__ vT,
    const float* __restrict__ pe, const float* __restrict__ pmask,
    const _Float16* __restrict__ wpairT, const float* __restrict__ csums,
    float* __restrict__ opart,   // [S][B][H][I][C]
    float* __restrict__ mpart,   // [S][B][H][I]
    float* __restrict__ lpart) { // [S][B][H][I]
  const int i0 = blockIdx.x * 16;
  const int b  = blockIdx.y;
  const int js = blockIdx.z;
  const int w  = threadIdx.x >> 5;       // wave index == head
  const int lane = threadIdx.x & 31;
  const int col = lane & 15, hf = lane >> 4;

  __shared__ float sbias[2][H_][16][32];                 // 32 KB, double-buffered
  __shared__ __align__(32) _Float16 pl[H_][16 * 32];     // 8 KB, per-wave P tiles

  const v8f zero = {};
  v16h qa  = load_a16(q16 + ((size_t)(b * I_ + i0)) * HC_ + w * C_, HC_);
  v16h wb0 = load_bn16(wpairT + 0,  P_);
  v16h wb1 = load_bn16(wpairT + 32, P_);
  const float colsum_c = csums[col];
  const float cbeta_c  = csums[16 + col];
  const _Float16* vbase = vT + ((size_t)(b * H_ + w)) * C_ * J_;

  float mrow[8], lrow[8];
  v8f oc0 = {}, oc1 = {};
#pragma unroll
  for (int r = 0; r < 8; ++r) { mrow[r] = -3.0e38f; lrow[r] = 0.f; }

  const int jbeg = js * JSEG;
  for (int j0 = jbeg, s = 0; j0 < jbeg + JSEG; j0 += 32, ++s) {
    float (*sb)[16][32] = sbias[s & 1];

    // ---------------- stage 1: bias tiles (4 A-tiles per wave) ----------------
#pragma unroll
    for (int tt = 0; tt < 4; ++tt) {
      const int t   = w * 4 + tt;          // 0..31
      const int il  = t >> 1;              // i row within tile
      const int jjb = (t & 1) * 16;        // jj half
      const float* per = pe + (((size_t)(b * I_ + i0 + il)) * J_ + j0 + jjb + col) * P_;
      float s1v = 0.f, s2v = 0.f;
      v16h a0, a1;
#pragma unroll
      for (int e = 0; e < 8; ++e) {
        const float v0 = per[8 * hf + e];
        const float v1 = per[16 + 8 * hf + e];
        const float v2 = per[32 + 8 * hf + e];
        const float v3 = per[48 + 8 * hf + e];
        a0[e] = (_Float16)v0; a0[8 + e] = (_Float16)v1;
        a1[e] = (_Float16)v2; a1[8 + e] = (_Float16)v3;
        s1v += v0 + v1 + v2 + v3;
        s2v += v0 * v0 + v1 * v1 + v2 * v2 + v3 * v3;
      }
      s1v += __shfl_xor(s1v, 16, 32);      // lanes m and m+16 hold row m halves
      s2v += __shfl_xor(s2v, 16, 32);
      const float mean = s1v * (1.f / P_);
      const float rstd = rsqrtf(s2v * (1.f / P_) - mean * mean + EPS_);
      v8f pacc = wmma_f16(a0, wb0, zero);
      pacc     = wmma_f16(a1, wb1, pacc);
      const float* pmr = pmask + ((size_t)(b * I_ + i0 + il)) * J_ + j0 + jjb + 8 * hf;
#pragma unroll
      for (int r = 0; r < 8; ++r) {
        const float mr = __shfl(mean, r + 8 * hf, 32);   // row-stat broadcast
        const float rr = __shfl(rstd, r + 8 * hf, 32);
        const float pn = rr * (pacc[r] - mr * colsum_c) + cbeta_c;
        const float pg = __shfl_xor(pn, 8, 32);          // pair (proj, gate)
        if (col < 8) {
          const float bv = pn * fast_sigmoid(pg) + INF_ * (pmr[r] - 1.f);
          sb[col][il][jjb + 8 * hf + r] = bv;            // h = col
        }
      }
    }
    __syncthreads();

    // ---------------- stage 2: attention step for head w ----------------
    const _Float16* kp = k16 + ((size_t)(b * J_ + j0)) * HC_ + w * C_;
    if (j0 + 32 < jbeg + JSEG) __builtin_prefetch(kp + 32 * HC_, 0, 1);
    v16h kb0 = load_bn16(kp, HC_);
    v16h kb1 = load_bn16(kp + 16 * HC_, HC_);
    v8f s0 = wmma_f16(qa, kb0, zero);
    v8f s1 = wmma_f16(qa, kb1, zero);
#pragma unroll
    for (int r = 0; r < 8; ++r) {
      s0[r] += sb[w][r + 8 * hf][col];
      s1[r] += sb[w][r + 8 * hf][16 + col];
    }
#pragma unroll
    for (int r = 0; r < 8; ++r) {
      float mx = fmaxf(s0[r], s1[r]);
#pragma unroll
      for (int m = 8; m >= 1; m >>= 1) mx = fmaxf(mx, __shfl_xor(mx, m, 32));
      const float mnew = fmaxf(mrow[r], mx);
      const float corr = __expf(mrow[r] - mnew);
      const float p0 = __expf(s0[r] - mnew);
      const float p1 = __expf(s1[r] - mnew);
      float rs = p0 + p1;
#pragma unroll
      for (int m = 8; m >= 1; m >>= 1) rs += __shfl_xor(rs, m, 32);
      lrow[r] = lrow[r] * corr + rs;
      mrow[r] = mnew;
      oc0[r] *= corr;
      oc1[r] *= corr;
      pl[w][(r + 8 * hf) * 32 + col]      = (_Float16)p0;
      pl[w][(r + 8 * hf) * 32 + 16 + col] = (_Float16)p1;
    }
    asm volatile("s_wait_dscnt 0" ::: "memory");
    v16h pa  = load_a16(pl[w], 32);        // C-layout -> A-layout via LDS
    v16h vb0 = load_bn16(vbase + (size_t)0  * J_ + j0, J_);
    v16h vb1 = load_bn16(vbase + (size_t)16 * J_ + j0, J_);
    oc0 = wmma_f16(pa, vb0, oc0);
    oc1 = wmma_f16(pa, vb1, oc1);
  }

  // emit split partials
  float* ob = opart + ((((size_t)js * B_ + b) * H_ + w) * I_ + i0) * C_;
  float* mb = mpart + (((size_t)js * B_ + b) * H_ + w) * I_ + i0;
  float* lb = lpart + (((size_t)js * B_ + b) * H_ + w) * I_ + i0;
#pragma unroll
  for (int r = 0; r < 8; ++r) {
    const int row = r + 8 * hf;
    ob[(size_t)row * C_ + col]      = oc0[r];
    ob[(size_t)row * C_ + 16 + col] = oc1[r];
    if (col == 0) { mb[row] = mrow[r]; lb[row] = lrow[r]; }
  }
}

// ---------------------------------------------------------------------------
// Kernel 3b: merge split-J partials, normalize, apply gate, store o16 f16.
// One thread per (b,h,i,c); c fastest -> coalesced.
// ---------------------------------------------------------------------------
__global__ __launch_bounds__(256) void attn_merge_kernel(
    const float* __restrict__ opart, const float* __restrict__ mpart,
    const float* __restrict__ lpart, const _Float16* __restrict__ g16,
    _Float16* __restrict__ o16) {
  const int t = blockIdx.x * blockDim.x + threadIdx.x;   // B*H*I*C = 2^19
  const int c = t & 31;
  const int i = (t >> 5) & (I_ - 1);
  const int h = (t >> 15) & (H_ - 1);
  const int b = t >> 18;
  float ms[S_];
  float M = -3.0e38f;
#pragma unroll
  for (int s = 0; s < S_; ++s) {
    ms[s] = mpart[(((size_t)s * B_ + b) * H_ + h) * I_ + i];
    M = fmaxf(M, ms[s]);
  }
  float L = 0.f, O = 0.f;
#pragma unroll
  for (int s = 0; s < S_; ++s) {
    const float wgt = __expf(ms[s] - M);
    L += lpart[(((size_t)s * B_ + b) * H_ + h) * I_ + i] * wgt;
    O += opart[((((size_t)s * B_ + b) * H_ + h) * I_ + i) * C_ + c] * wgt;
  }
  const size_t oidx = ((size_t)(b * I_ + i)) * HC_ + h * C_ + c;
  const float g = (float)g16[oidx];
  o16[oidx] = (_Float16)(O * __builtin_amdgcn_rcpf(L) * g);
}

// ---------------------------------------------------------------------------
// Kernel 4: out projection [2048 x 256] @ [256 x 512] + bo, row mask, residual.
// ---------------------------------------------------------------------------
__global__ __launch_bounds__(256) void out_proj_kernel(
    const _Float16* __restrict__ o16, const _Float16* __restrict__ woT16,
    const float* __restrict__ bo, const float* __restrict__ ni32,
    const float* __restrict__ nmask, float* __restrict__ out) {
  const int wave = (blockIdx.x * blockDim.x + threadIdx.x) >> 5;
  const int ntl  = D_ / 16;
  const int m0   = (wave / ntl) * 16;
  const int n0   = (wave % ntl) * 16;
  v8f acc = {};
#pragma unroll 4
  for (int k0 = 0; k0 < HC_; k0 += 32) {
    v16h a = load_a16(o16 + (size_t)m0 * HC_ + k0, HC_);
    v16h b = load_bn16(woT16 + (size_t)n0 * HC_ + k0, HC_);
    acc = wmma_f16(a, b, acc);
  }
  const int lane = threadIdx.x & 31;
  const int col = lane & 15, hf = lane >> 4;
#pragma unroll
  for (int r = 0; r < 8; ++r) {
    const int row = m0 + r + 8 * hf;
    const float mha = acc[r] + bo[n0 + col];
    const float res = ni32[(size_t)row * D_ + n0 + col];
    out[(size_t)row * D_ + n0 + col] = res + mha * nmask[row];
  }
}

// ---------------------------------------------------------------------------
extern "C" void kernel_launch(void* const* d_in, const int* in_sizes, int n_in,
                              void* d_out, int out_size, void* d_ws, size_t ws_size,
                              hipStream_t stream) {
  const float* node_i = (const float*)d_in[0];
  const float* node_j = (const float*)d_in[1];
  const float* pe     = (const float*)d_in[2];
  const float* pmask  = (const float*)d_in[3];
  const float* nmask  = (const float*)d_in[4];
  const float* ln_i_g = (const float*)d_in[5];
  const float* ln_i_b = (const float*)d_in[6];
  const float* ln_j_g = (const float*)d_in[7];
  const float* ln_j_b = (const float*)d_in[8];
  const float* ln_p_g = (const float*)d_in[9];
  const float* ln_p_b = (const float*)d_in[10];
  const float* Wp     = (const float*)d_in[11];
  const float* bp     = (const float*)d_in[12];
  const float* Wpg    = (const float*)d_in[13];
  const float* bpg    = (const float*)d_in[14];
  const float* Wq     = (const float*)d_in[15];
  const float* Wk     = (const float*)d_in[16];
  const float* Wv     = (const float*)d_in[17];
  const float* Wg     = (const float*)d_in[18];
  const float* bg     = (const float*)d_in[19];
  const float* Wo     = (const float*)d_in[20];
  const float* bo     = (const float*)d_in[21];
  float* out = (float*)d_out;

  // workspace carving (all offsets >=256B-aligned)
  char* ws = (char*)d_ws;
  size_t off = 0;
  float*    ni32  = (float*)(ws + off);    off += sizeof(float)    * 2048ull * D_;
  _Float16* ni16  = (_Float16*)(ws + off); off += sizeof(_Float16) * 2048ull * D_;
  _Float16* nj16  = (_Float16*)(ws + off); off += sizeof(_Float16) * 2048ull * D_;
  _Float16* qkvg  = (_Float16*)(ws + off); off += sizeof(_Float16) * 4ull * 2048 * HC_;
  _Float16* o16   = (_Float16*)(ws + off); off += sizeof(_Float16) * 2048ull * HC_;
  _Float16* wT16  = (_Float16*)(ws + off); off += sizeof(_Float16) * 4ull * D_ * HC_;
  _Float16* woT   = (_Float16*)(ws + off); off += sizeof(_Float16) * (size_t)HC_ * D_;
  _Float16* wpairT= (_Float16*)(ws + off); off += 4096;   // 16*64 f16 = 2 KB (+pad)
  float*    csums = (float*)(ws + off);    off += 256;
  float*    opart = (float*)(ws + off);    off += sizeof(float) * (size_t)S_ * B_ * H_ * I_ * C_;
  float*    mpart = (float*)(ws + off);    off += sizeof(float) * (size_t)S_ * B_ * H_ * I_;
  float*    lpart = (float*)(ws + off);    off += sizeof(float) * (size_t)S_ * B_ * H_ * I_;
  (void)ws_size; (void)in_sizes; (void)n_in; (void)out_size;

  _Float16* q16 = qkvg + 0ull * 2048 * HC_;
  _Float16* k16 = qkvg + 1ull * 2048 * HC_;
  _Float16* vT  = qkvg + 2ull * 2048 * HC_;  // reinterpreted as [B][H][C][J]
  _Float16* g16 = qkvg + 3ull * 2048 * HC_;

  const int WN = D_ * HC_;
  // 0) weight prep
  cvtT_f16_kernel<<<WN / 256, 256, 0, stream>>>(Wq, wT16 + 0ull * WN, D_, HC_);
  cvtT_f16_kernel<<<WN / 256, 256, 0, stream>>>(Wk, wT16 + 1ull * WN, D_, HC_);
  cvtT_f16_kernel<<<WN / 256, 256, 0, stream>>>(Wv, wT16 + 2ull * WN, D_, HC_);
  cvtT_f16_kernel<<<WN / 256, 256, 0, stream>>>(Wg, wT16 + 3ull * WN, D_, HC_);
  cvtT_f16_kernel<<<WN / 256, 256, 0, stream>>>(Wo, woT, HC_, D_);
  prep_pair_kernel<<<1, 256, 0, stream>>>(Wp, bp, Wpg, bpg, ln_p_g, ln_p_b, wpairT, csums);

  // 1) node layernorms
  ln_nodes_kernel<<<512, 256, 0, stream>>>(node_i, node_j, ln_i_g, ln_i_b,
                                           ln_j_g, ln_j_b, ni32, ni16, nj16);

  // 2) q/k/v/g projections
  qkvg_gemm_kernel<<<dim3(256, 4, 1), 256, 0, stream>>>(ni16, nj16, wT16, bg, qkvg, vT);

  // 3) fused pair-bias + flash attention, split-J: 64*2*4 = 512 blocks
  attn_fused_kernel<<<dim3(I_ / 16, B_, S_), 256, 0, stream>>>(
      q16, k16, vT, pe, pmask, wpairT, csums, opart, mpart, lpart);

  // 3b) merge split partials + gate
  attn_merge_kernel<<<(B_ * H_ * I_ * C_) / 256, 256, 0, stream>>>(
      opart, mpart, lpart, g16, o16);

  // 4) out projection + gate-residual
  out_proj_kernel<<<512, 256, 0, stream>>>(o16, woT, bo, ni32, nmask, out);
}
